// PinnWithGRU_82514911691403
// MI455X (gfx1250) — matrix operand order repaired and checked by reading, not compiled, not run on hardware
//
#include <hip/hip_runtime.h>
#include <hip/hip_bf16.h>
#include <stdint.h>

// GRU: T=1024 sequential steps, B=64, H=2048, I=3.
// Per step: hg = h @ W_hh^T  ([64x2048]*[2048x6144], bf16 WMMA, f32 accum),
// then elementwise gates, then device-wide barrier (ping-pong h buffers).

typedef __attribute__((ext_vector_type(16))) __bf16 v16bf;
typedef __attribute__((ext_vector_type(8)))  float  v8f;
typedef __attribute__((ext_vector_type(4)))  unsigned int u32x4;

#define T_STEPS 1024
#define BATCH   64
#define HID     2048
#define G3      (3 * HID)
#define NWG     (HID / 16)   // 128 workgroups, each owns 16 hidden columns
#define WGSIZE  128          // 4 waves; wave w handles batch rows 16w..16w+15

union Frag { u32x4 q[2]; v16bf v; };

__device__ __forceinline__ unsigned short f2bf(float f) {
  unsigned int u = __builtin_bit_cast(unsigned int, f);
  u += 0x7FFFu + ((u >> 16) & 1u);          // round to nearest even
  return (unsigned short)(u >> 16);
}
__device__ __forceinline__ float sigm(float x)  { return 1.0f / (1.0f + __expf(-x)); }
__device__ __forceinline__ float tanh_(float x) { return 1.0f - 2.0f / (1.0f + __expf(2.0f * x)); }

// One-time (per launch) setup: zero barrier counter, W_hh fp32->bf16 (L2-resident
// thereafter), init hidden -> bf16 ping buffer.
__global__ void gru_init(const float* __restrict__ Whh,
                         const float* __restrict__ h0,
                         unsigned short* __restrict__ Wbf,
                         unsigned short* __restrict__ hbf0,
                         unsigned int* __restrict__ cnt)
{
  size_t i = (size_t)blockIdx.x * blockDim.x + threadIdx.x;
  size_t stride = (size_t)gridDim.x * blockDim.x;
  if (i == 0) *cnt = 0u;
  for (size_t k = i; k < (size_t)G3 * HID; k += stride)   Wbf[k]  = f2bf(Whh[k]);
  for (size_t k = i; k < (size_t)BATCH * HID; k += stride) hbf0[k] = f2bf(h0[k]);
}

__global__ __launch_bounds__(WGSIZE, 1)
void gru_persistent(const float* __restrict__ inp,
                    const float* __restrict__ h0,
                    const float* __restrict__ Wih,
                    const float* __restrict__ bias,
                    const float* __restrict__ bias_n,
                    const unsigned short* __restrict__ Wbf,
                    unsigned short* __restrict__ hbf0,
                    unsigned short* __restrict__ hbf1,
                    unsigned int* __restrict__ cnt,
                    float* __restrict__ out)
{
  const int tid  = threadIdx.x;
  const int wave = tid >> 5;
  const int lane = tid & 31;
  const int half = lane >> 4;
  const int l16  = lane & 15;

  const int j    = blockIdx.x * 16 + l16;   // hidden column owned by this lane
  const int brow = wave * 16 + half * 8;    // first C-tile batch row for this lane

  // Hoist time-invariant input-projection weights/biases for column j.
  const float wr0 = Wih[(size_t)j * 3 + 0],           wr1 = Wih[(size_t)j * 3 + 1],           wr2 = Wih[(size_t)j * 3 + 2];
  const float wz0 = Wih[(size_t)(HID + j) * 3 + 0],   wz1 = Wih[(size_t)(HID + j) * 3 + 1],   wz2 = Wih[(size_t)(HID + j) * 3 + 2];
  const float wn0 = Wih[(size_t)(2*HID + j) * 3 + 0], wn1 = Wih[(size_t)(2*HID + j) * 3 + 1], wn2 = Wih[(size_t)(2*HID + j) * 3 + 2];
  const float br_ = bias[j], bz_ = bias[HID + j], bn_ = bias[2*HID + j], bnn = bias_n[j];

  // h_old lives in registers: exactly the elements this lane produces each step.
  float hold[8];
  #pragma unroll
  for (int v = 0; v < 8; ++v) hold[v] = h0[(size_t)(brow + v) * HID + j];

  // B fragments (W_hh^T columns = W_hh rows): per-lane 32 contiguous bf16 bytes in K.
  const unsigned short* pr = Wbf + (size_t)j * HID             + half * 16;
  const unsigned short* pz = Wbf + (size_t)(HID   + j) * HID   + half * 16;
  const unsigned short* pn = Wbf + (size_t)(2*HID + j) * HID   + half * 16;
  // A fragment row base: per-lane 32 contiguous bf16 bytes in K (matches B).
  const size_t arow = (size_t)(wave * 16 + l16) * HID + half * 16;

  const bool writer = (blockIdx.x == 0) && (l16 == 0);  // owns output column j==0
  const unsigned int nblk = gridDim.x;

  for (int t = 0; t < T_STEPS; ++t) {
    const unsigned short* hR = (t & 1) ? hbf1 : hbf0;   // ping-pong h buffers
    unsigned short*       hW = (t & 1) ? hbf0 : hbf1;
    const unsigned short* pa = hR + arow;

    v8f accr = {}, accz = {}, accn = {};
    for (int kk = 0; kk < HID; kk += 32) {
      Frag a, fr, fz, fn;
      a.q[0]  = *(const u32x4*)(pa + kk);        // K = kk + 16*half .. +7
      a.q[1]  = *(const u32x4*)(pa + kk + 8);    // K = kk+8 + 16*half .. +7
      fr.q[0] = *(const u32x4*)(pr + kk);
      fr.q[1] = *(const u32x4*)(pr + kk + 8);
      fz.q[0] = *(const u32x4*)(pz + kk);
      fz.q[1] = *(const u32x4*)(pz + kk + 8);
      fn.q[0] = *(const u32x4*)(pn + kk);
      fn.q[1] = *(const u32x4*)(pn + kk + 8);
      accr = __builtin_amdgcn_wmma_f32_16x16x32_bf16(false, a.v, false, fr.v, (short)0, accr, false, false);
      accz = __builtin_amdgcn_wmma_f32_16x16x32_bf16(false, a.v, false, fz.v, (short)0, accz, false, false);
      accn = __builtin_amdgcn_wmma_f32_16x16x32_bf16(false, a.v, false, fn.v, (short)0, accn, false, false);
    }

    // Gate epilogue in f32 on C-tile layout: lane holds (b=brow+v, col=j).
    #pragma unroll
    for (int v = 0; v < 8; ++v) {
      const int b = brow + v;
      const float* ip = inp + ((size_t)t * BATCH + b) * 3;
      const float x0 = ip[0], x1 = ip[1], x2 = ip[2];
      const float r  = sigm(x0*wr0 + x1*wr1 + x2*wr2 + br_ + accr[v]);
      const float z  = sigm(x0*wz0 + x1*wz1 + x2*wz2 + bz_ + accz[v]);
      const float nw = tanh_(x0*wn0 + x1*wn1 + x2*wn2 + bn_ + r * (accn[v] + bnn));
      const float hn = nw + z * (hold[v] - nw);
      hold[v] = hn;
      hW[(size_t)b * HID + j] = f2bf(hn);
      if (writer) out[(size_t)t * BATCH + b] = hn;   // out[t] = h_new[:, 0]
    }

    // Device-wide step barrier: monotonic counter, release/acquire at agent scope.
    __syncthreads();
    if (tid == 0) {
      __hip_atomic_fetch_add(cnt, 1u, __ATOMIC_RELEASE, __HIP_MEMORY_SCOPE_AGENT);
      const unsigned int target = nblk * (unsigned int)(t + 1);
      while (__hip_atomic_load(cnt, __ATOMIC_ACQUIRE, __HIP_MEMORY_SCOPE_AGENT) < target)
        __builtin_amdgcn_s_sleep(1);
    }
    __syncthreads();
  }
}

extern "C" void kernel_launch(void* const* d_in, const int* in_sizes, int n_in,
                              void* d_out, int out_size, void* d_ws, size_t ws_size,
                              hipStream_t stream) {
  (void)in_sizes; (void)n_in; (void)out_size; (void)ws_size;
  const float* inp  = (const float*)d_in[0];  // [T,B,I]
  const float* h0   = (const float*)d_in[1];  // [B,H]
  const float* Wih  = (const float*)d_in[2];  // [3H,I]
  const float* Whh  = (const float*)d_in[3];  // [3H,H]
  const float* bias = (const float*)d_in[4];  // [3H]
  const float* bn   = (const float*)d_in[5];  // [H]
  float* out = (float*)d_out;                 // [T,B]

  char* ws = (char*)d_ws;
  unsigned int*   cnt  = (unsigned int*)ws;
  unsigned short* hbf0 = (unsigned short*)(ws + 256);
  unsigned short* hbf1 = hbf0 + (size_t)BATCH * HID;
  unsigned short* Wbf  = hbf1 + (size_t)BATCH * HID;   // 24 MB bf16 weights (L2-resident)

  gru_init<<<4096, 256, 0, stream>>>(Whh, h0, Wbf, hbf0, cnt);
  gru_persistent<<<NWG, WGSIZE, 0, stream>>>(inp, h0, Wih, bias, bn, Wbf, hbf0, hbf1, cnt, out);
}
